// EnsembleModel_36232344109238
// MI455X (gfx1250) — matrix operand order repaired and checked by reading, not compiled, not run on hardware
//
#include <hip/hip_runtime.h>
#include <hip/hip_bf16.h>

typedef __attribute__((ext_vector_type(16))) _Float16 v16h;
typedef __attribute__((ext_vector_type(8)))  _Float16 v8h;
typedef __attribute__((ext_vector_type(8)))  float    v8f;
typedef __attribute__((ext_vector_type(4)))  float    v4f;

#define NE   10
#define NIN  32
#define NH   256
#define NOUT 24
#define NO2  48
#define NB   65536

// LDS layout (bytes). Dynamic shared memory, ~298 KB total (<320KB WGP LDS).
#define W2F_OFF   0        // 16 ntiles * 8 kblocks * 512 halves = 131072 B
#define W1F_OFF   131072   // 16 ntiles * 1 kblock  * 512 halves = 16384 B
#define W3F_OFF   147456   //  3 ntiles * 8 kblocks * 512 halves = 24576 B
#define HBUF_OFF  172032   //  8 waves * 32 rows * 256 cols f16  = 131072 B
#define B1_OFF    303104   // 256 f32
#define B2_OFF    304128   // 256 f32
#define B3_OFF    305152   // 48 f32
#define SMEM_BYTES 305408

__device__ __forceinline__ float silu_f(float v) {
    return v / (1.0f + __expf(-v));
}

__device__ __forceinline__ v16h cat8(v8h lo, v8h hi) {
    return __builtin_shufflevector(lo, hi, 0,1,2,3,4,5,6,7,8,9,10,11,12,13,14,15);
}

// A-fragment (16x32 f16) from a row-major f16 LDS tile: lane row = lane&15,
// halves 0..7 = K(kb*32+8g .. +7), halves 8..15 = K(kb*32+16+8g .. +7).
__device__ __forceinline__ v16h lds_afrag(const _Float16* hb, int row0, int m,
                                          int kb, int g) {
    const _Float16* hrow = hb + (row0 + m) * NH + kb * 32 + 8 * g;
    v8h lo = *(const v8h*)(hrow);        // ds_load_b128
    v8h hi = *(const v8h*)(hrow + 16);   // ds_load_b128
    return cat8(lo, hi);
}

// 256 threads = 8 wave32; require 2 waves per SIMD32 to be allocable so the
// 8-wave workgroup can always be placed on a WGP (compiler caps VGPRs at RF/2
// and avoids the extended-VGPR s_set_vgpr_msb path).
__global__ void __launch_bounds__(256, 2)
ensemble_mlp_kernel(const float* __restrict__ x,
                    const float* __restrict__ W1, const float* __restrict__ b1,
                    const float* __restrict__ W2, const float* __restrict__ b2,
                    const float* __restrict__ W3, const float* __restrict__ b3,
                    float* __restrict__ out)
{
    extern __shared__ __align__(16) char smem[];
    _Float16* w2f    = (_Float16*)(smem + W2F_OFF);
    _Float16* w1f    = (_Float16*)(smem + W1F_OFF);
    _Float16* w3f    = (_Float16*)(smem + W3F_OFF);
    _Float16* hb_all = (_Float16*)(smem + HBUF_OFF);
    float* b1s = (float*)(smem + B1_OFF);
    float* b2s = (float*)(smem + B2_OFF);
    float* b3s = (float*)(smem + B3_OFF);

    const int tid  = threadIdx.x;
    const int e    = blockIdx.y;
    const int lane = tid & 31;
    const int wid  = tid >> 5;
    const int g    = lane >> 4;   // lane group (WMMA half-wave)
    const int ln   = lane & 15;

    // ---- Stage expert weights: f32 global -> f16 LDS in WMMA B-fragment order ----
    // B-fragment (32x16 K-major tile): lane holds N = nt*16 + (lane&15);
    // half h maps to K = kb*32 + 8*(lane>>4) + h (h<8), kb*32 + 16 + 8*(lane>>4) + h-8 (h>=8). Store [frag][lane][half] so each
    // lane's operand is one contiguous 32B LDS read.
    const float* W1e = W1 + (size_t)e * NIN * NH;
    const float* W2e = W2 + (size_t)e * NH  * NH;
    const float* W3e = W3 + (size_t)e * NH  * NO2;

    for (int idx = tid; idx < NH * NH; idx += 256) {          // W2: coalesced source
        int K = idx >> 8, N = idx & 255;
        int kb = K >> 5, kk = K & 31;
        int gg = (kk >> 3) & 1, hh = (kk & 7) | ((kk >> 4) << 3);
        int nt = N >> 4, nn = N & 15;
        int d = (((nt << 3) + kb) << 9) + ((gg * 16 + nn) << 4) + hh;
        w2f[d] = (_Float16)W2e[idx];
    }
    for (int idx = tid; idx < NIN * NH; idx += 256) {         // W1
        int K = idx >> 8, N = idx & 255;
        int gg = (K >> 3) & 1, hh = (K & 7) | ((K >> 4) << 3);
        int nt = N >> 4, nn = N & 15;
        int d = (nt << 9) + ((gg * 16 + nn) << 4) + hh;
        w1f[d] = (_Float16)W1e[idx];
    }
    for (int idx = tid; idx < NH * NO2; idx += 256) {         // W3
        int K = idx / NO2, N = idx % NO2;
        int kb = K >> 5, kk = K & 31;
        int gg = (kk >> 3) & 1, hh = (kk & 7) | ((kk >> 4) << 3);
        int nt = N >> 4, nn = N & 15;
        int d = (((nt << 3) + kb) << 9) + ((gg * 16 + nn) << 4) + hh;
        w3f[d] = (_Float16)W3e[idx];
    }
    for (int d = tid; d < NH; d += 256) {
        b1s[d] = b1[(size_t)e * NH + d];
        b2s[d] = b2[(size_t)e * NH + d];
    }
    if (tid < NO2) b3s[tid] = b3[(size_t)e * NO2 + tid];
    __syncthreads();

    _Float16* hb = hb_all + wid * (32 * NH);        // this wave's 32x256 f16 buffer
    const int rowBase = blockIdx.x * 256 + wid * 32;
    const int m = ln;                               // A-fragment batch row for lane

    // ---- Layer 1: h1 = silu(x @ W1 + b1), K = 32; two row tiles share each B ----
    {
        v16h a0, a1;
        #pragma unroll
        for (int rt = 0; rt < 2; ++rt) {
            const float* xr = x + (size_t)(rowBase + rt * 16 + m) * NIN;
            v4f x0 = *(const v4f*)(xr + 8 * g);
            v4f x1 = *(const v4f*)(xr + 8 * g + 4);
            v4f x2 = *(const v4f*)(xr + 16 + 8 * g);
            v4f x3 = *(const v4f*)(xr + 16 + 8 * g + 4);
            v16h a;
            #pragma unroll
            for (int t = 0; t < 4; ++t) {
                a[t]      = (_Float16)x0[t];
                a[t + 4]  = (_Float16)x1[t];
                a[t + 8]  = (_Float16)x2[t];
                a[t + 12] = (_Float16)x3[t];
            }
            if (rt == 0) a0 = a; else a1 = a;
        }
        #pragma unroll
        for (int nt = 0; nt < 16; ++nt) {
            v16h bfr = *(const v16h*)(w1f + (nt << 9) + lane * 16);
            v8f c0 = {}, c1 = {};
            c0 = __builtin_amdgcn_wmma_f32_16x16x32_f16(false, a0, false, bfr,
                                                        (short)0, c0, false, false);
            c1 = __builtin_amdgcn_wmma_f32_16x16x32_f16(false, a1, false, bfr,
                                                        (short)0, c1, false, false);
            float bias = b1s[nt * 16 + ln];
            #pragma unroll
            for (int i = 0; i < 8; ++i) {   // C layout: row = i + 8g, col = nt*16+ln
                hb[(i + 8 * g) * NH + nt * 16 + ln]      = (_Float16)silu_f(c0[i] + bias);
                hb[(16 + i + 8 * g) * NH + nt * 16 + ln] = (_Float16)silu_f(c1[i] + bias);
            }
        }
    }

    // ---- Layer 2: h2 = silu(h1 @ W2 + b2), 32x256 @ 256x256 ----
    // Two N-passes of 8 tiles each; pass-0 results are held in registers (packed
    // f16) and flushed only after pass 1 has finished reading h1 (WAR hazard).
    {
        v8h hold0[8], hold1[8];
        v8f zero = {};
        #pragma unroll
        for (int half = 0; half < 2; ++half) {
            v8f acc0[8], acc1[8];
            #pragma unroll
            for (int t = 0; t < 8; ++t) { acc0[t] = zero; acc1[t] = zero; }

            #pragma unroll
            for (int kb = 0; kb < 8; ++kb) {
                v16h a0 = lds_afrag(hb, 0,  m, kb, g);
                v16h a1 = lds_afrag(hb, 16, m, kb, g);
                #pragma unroll
                for (int t = 0; t < 8; ++t) {
                    int nt = half * 8 + t;
                    v16h bfr = *(const v16h*)(w2f + (((nt << 3) + kb) << 9) + lane * 16);
                    acc0[t] = __builtin_amdgcn_wmma_f32_16x16x32_f16(false, a0, false, bfr,
                                                                     (short)0, acc0[t], false, false);
                    acc1[t] = __builtin_amdgcn_wmma_f32_16x16x32_f16(false, a1, false, bfr,
                                                                     (short)0, acc1[t], false, false);
                }
            }

            if (half == 0) {
                #pragma unroll
                for (int t = 0; t < 8; ++t) {
                    float bias = b2s[t * 16 + ln];
                    #pragma unroll
                    for (int i = 0; i < 8; ++i) {
                        hold0[t][i] = (_Float16)silu_f(acc0[t][i] + bias);
                        hold1[t][i] = (_Float16)silu_f(acc1[t][i] + bias);
                    }
                }
            } else {
                // All h1 reads are complete: write pass-1 results...
                #pragma unroll
                for (int t = 0; t < 8; ++t) {
                    float bias = b2s[128 + t * 16 + ln];
                    int col = 128 + t * 16 + ln;
                    #pragma unroll
                    for (int i = 0; i < 8; ++i) {
                        hb[(i + 8 * g) * NH + col]      = (_Float16)silu_f(acc0[t][i] + bias);
                        hb[(16 + i + 8 * g) * NH + col] = (_Float16)silu_f(acc1[t][i] + bias);
                    }
                }
                // ...and flush the held pass-0 results.
                #pragma unroll
                for (int t = 0; t < 8; ++t) {
                    int col = t * 16 + ln;
                    #pragma unroll
                    for (int i = 0; i < 8; ++i) {
                        hb[(i + 8 * g) * NH + col]      = hold0[t][i];
                        hb[(16 + i + 8 * g) * NH + col] = hold1[t][i];
                    }
                }
            }
        }
    }

    // ---- Layer 3: pred = h2 @ W3 + b3, N = 48 (3 tiles), B shared by 2 row tiles ----
    {
        v8f zero = {};
        v8f acc0[3], acc1[3];
        #pragma unroll
        for (int nt = 0; nt < 3; ++nt) { acc0[nt] = zero; acc1[nt] = zero; }

        #pragma unroll
        for (int kb = 0; kb < 8; ++kb) {
            v16h a0 = lds_afrag(hb, 0,  m, kb, g);
            v16h a1 = lds_afrag(hb, 16, m, kb, g);
            #pragma unroll
            for (int nt = 0; nt < 3; ++nt) {
                v16h bfr = *(const v16h*)(w3f + (((nt << 3) + kb) << 9) + lane * 16);
                acc0[nt] = __builtin_amdgcn_wmma_f32_16x16x32_f16(false, a0, false, bfr,
                                                                  (short)0, acc0[nt], false, false);
                acc1[nt] = __builtin_amdgcn_wmma_f32_16x16x32_f16(false, a1, false, bfr,
                                                                  (short)0, acc1[nt], false, false);
            }
        }
        const size_t lvBase = (size_t)NE * NB * NOUT;   // start of logvars
        #pragma unroll
        for (int nt = 0; nt < 3; ++nt) {
            int o = nt * 16 + ln;                        // output feature 0..47
            float bias = b3s[o];
            #pragma unroll
            for (int rt = 0; rt < 2; ++rt) {
                #pragma unroll
                for (int i = 0; i < 8; ++i) {
                    int row = rowBase + rt * 16 + i + 8 * g;
                    float v = (rt == 0 ? acc0[nt][i] : acc1[nt][i]) + bias;
                    size_t idx = (o < NOUT)
                        ? ((size_t)(e * NB + row) * NOUT + o)
                        : (lvBase + (size_t)(e * NB + row) * NOUT + (o - NOUT));
                    out[idx] = v;
                }
            }
        }
    }
}

extern "C" void kernel_launch(void* const* d_in, const int* in_sizes, int n_in,
                              void* d_out, int out_size, void* d_ws, size_t ws_size,
                              hipStream_t stream) {
    const float* x  = (const float*)d_in[0];
    const float* W1 = (const float*)d_in[1];
    const float* b1 = (const float*)d_in[2];
    const float* W2 = (const float*)d_in[3];
    const float* b2 = (const float*)d_in[4];
    const float* W3 = (const float*)d_in[5];
    const float* b3 = (const float*)d_in[6];
    float* out = (float*)d_out;

    (void)in_sizes; (void)n_in; (void)out_size; (void)d_ws; (void)ws_size;

    // Opt in to >64KB dynamic LDS (CDNA5 WGP has 320KB). Idempotent, capture-safe.
    hipFuncSetAttribute(reinterpret_cast<const void*>(&ensemble_mlp_kernel),
                        hipFuncAttributeMaxDynamicSharedMemorySize, SMEM_BYTES);

    dim3 grid(NB / 256, NE);   // (256, 10): 8 waves/block, 32 batch rows per wave
    ensemble_mlp_kernel<<<grid, 256, SMEM_BYTES, stream>>>(x, W1, b1, W2, b2, W3, b3, out);
}
